// HNetEnergyViaBoolWeights_42271068127505
// MI455X (gfx1250) — 8-bit matrix operand order repaired and checked by reading, compiled, not run on hardware
//
#include <hip/hip_runtime.h>

#define N_PTS    512
#define N_NODES  10000
#define N_EDGES  50000
#define N_CMP    64
#define KCOLS    250000
#define E_PAD    50048
#define KTILE    64
#define MTILE    128
#define N_KTILES (E_PAD / KTILE)
#define N_KCHUNK 64
#define OUT_ELTS (N_PTS * N_CMP)
#define FP8_ONE  0x38u
#define A_SCALE  512.0f

#define LDS_ROW   144
#define LDS_TILE  (128 * LDS_ROW)

typedef int   v16i __attribute__((ext_vector_type(16)));
typedef int   v4i  __attribute__((ext_vector_type(4)));
typedef int   v2i  __attribute__((ext_vector_type(2)));
typedef float v8f  __attribute__((ext_vector_type(8)));

typedef __attribute__((address_space(1))) v4i as1_v4i;
typedef __attribute__((address_space(3))) v4i as3_v4i;

#if defined(__has_builtin)
#if __has_builtin(__builtin_amdgcn_global_load_async_to_lds_b128)
#define HNET_HAVE_ASYNC 1
#else
#define HNET_HAVE_ASYNC 0
#endif
#else
#define HNET_HAVE_ASYNC 0
#endif

#define WS_TYPE8 ((size_t)0)
#define WS_B8    ((size_t)N_PTS * E_PAD)
#define WS_ACC   (WS_B8 + (size_t)N_KTILES * 128 * 128)
#define WS_MIN   (WS_ACC + (size_t)OUT_ELTS * sizeof(float))

__global__ void hnet_init_kernel(float* __restrict__ acc, int* __restrict__ minslot) {
    int i = blockIdx.x * blockDim.x + threadIdx.x;
    if (i < OUT_ELTS) acc[i] = 0.0f;
    if (i == 0) *minslot = 0x7F7FFFFF;
}

__global__ __launch_bounds__(256)
void hnet_types_kernel(const int* __restrict__ act,
                       const int* __restrict__ eidx,
                       unsigned char* __restrict__ type8) {
    const int E4 = E_PAD / 4;
    int gid = blockIdx.x * blockDim.x + threadIdx.x;
    int n  = gid / E4;
    int e4 = (gid - n * E4) * 4;
    if (n >= N_PTS) return;
    unsigned int w = 0u;
#pragma unroll
    for (int j = 0; j < 4; ++j) {
        int e = e4 + j;
        unsigned int t = 0u;
        if (e < N_EDGES) {
            int i0 = eidx[2 * e + 0];
            int i1 = eidx[2 * e + 1];
            int a  = act[n * N_NODES + i0];
            int b  = act[n * N_NODES + i1];
            t = 1u << (2 * a + b);
        }
        w |= t << (8 * j);
    }
    *(unsigned int*)(type8 + (size_t)n * E_PAD + e4) = w;
}

__global__ __launch_bounds__(256)
void hnet_prepb_kernel(const float* __restrict__ B,
                       unsigned char* __restrict__ b8) {
    const int total4 = N_KTILES * 4096;
    int gid = blockIdx.x * blockDim.x + threadIdx.x;
    if (gid >= total4) return;
    int j4   = (gid & 31) * 4;
    int row  = (gid >> 5) & 127;
    int tile = gid >> 12;
    int p = row >> 6, c = row & 63;
    unsigned int w = 0u;
#pragma unroll
    for (int j = 0; j < 4; ++j) {
        int jj = j4 + j;
        int ts = 2 * p + (jj >> 6);
        int e  = tile * KTILE + (jj & 63);
        if (e < N_EDGES) {
            float v = B[(size_t)c * KCOLS + (size_t)(ts + 1) * N_EDGES + e];
            if (v > 0.5f) w |= FP8_ONE << (8 * j);
        }
    }
    *(unsigned int*)(b8 + (size_t)gid * 4) = w;
}

__device__ __forceinline__ void hnet_stage_tile(const unsigned char* gsrc,
                                                unsigned char* ldst, int tid) {
#pragma unroll
    for (int k = 0; k < 4; ++k) {
        const int u    = tid * 4 + k;
        const int ldso = (u >> 3) * LDS_ROW + (u & 7) * 16;
        const unsigned char* g = gsrc + u * 16;
#if HNET_HAVE_ASYNC
        __builtin_amdgcn_global_load_async_to_lds_b128(
            (as1_v4i*)(unsigned long long)g,
            (as3_v4i*)(ldst + ldso),
            0, 0);
#else
        *(v4i*)(ldst + ldso) = *(const v4i*)g;
#endif
    }
}

template <int N> __device__ __forceinline__ void hnet_wait_async() {
#if HNET_HAVE_ASYNC
#if __has_builtin(__builtin_amdgcn_s_wait_asynccnt)
    __builtin_amdgcn_s_wait_asynccnt(N);
#else
    asm volatile("s_wait_asynccnt %0" ::"i"(N) : "memory");
#endif
#endif
}

__global__ __launch_bounds__(256)
void hnet_wmma_kernel(const unsigned char* __restrict__ type8,
                      const unsigned char* __restrict__ b8,
                      float* __restrict__ acc,
                      int tiles_per_chunk) {
    __shared__ unsigned char lbuf[2][LDS_TILE];

    const int tid   = threadIdx.x;
    const int mblk  = blockIdx.x;
    const int kchnk = blockIdx.y;
    const int wave  = tid >> 5;
    const int lane  = tid & 31;
    const int l15   = lane & 15;
    const int khalf = lane >> 4;

    const int mrow = mblk * MTILE + wave * 16 + l15;

    v8f acc0 = {}, acc1 = {}, acc2 = {}, acc3 = {};

    const int t0     = kchnk * tiles_per_chunk;
    const int remain = N_KTILES - t0;
    const int ntiles = remain < 0 ? 0 : (remain < tiles_per_chunk ? remain
                                                                  : tiles_per_chunk);

    const unsigned char* arow  = type8 + (size_t)mrow * E_PAD + khalf * 16
                               + (size_t)t0 * KTILE;
    const unsigned char* btile = b8 + (size_t)t0 * 16384;

    if (ntiles > 0) hnet_stage_tile(btile, &lbuf[0][0], tid);

    for (int tt = 0; tt < ntiles; ++tt) {
        const int cur = tt & 1;
        if (tt + 1 < ntiles) {
            hnet_stage_tile(btile + (size_t)(tt + 1) * 16384,
                            &lbuf[cur ^ 1][0], tid);
            hnet_wait_async<4>();
        } else {
            hnet_wait_async<0>();
        }
        __syncthreads();

        v2i q0 = *(const v2i*)(arow +  0);
        v2i q1 = *(const v2i*)(arow +  8);
        v2i q2 = *(const v2i*)(arow + 32);
        v2i q3 = *(const v2i*)(arow + 40);
        __builtin_prefetch(arow + KTILE, 0, 0);

        unsigned int r[8];
        r[0] = (unsigned int)q0[0]; r[1] = (unsigned int)q0[1];
        r[2] = (unsigned int)q1[0]; r[3] = (unsigned int)q1[1];
        r[4] = (unsigned int)q2[0]; r[5] = (unsigned int)q2[1];
        r[6] = (unsigned int)q3[0]; r[7] = (unsigned int)q3[1];

        const unsigned char* lbase = &lbuf[cur][0];

#pragma unroll
        for (int p = 0; p < 2; ++p) {
            v16i a;
#pragma unroll
            for (int i = 0; i < 8; ++i) {
                a[i]     = (int)((r[i] >> (2 * p))     & 0x01010101u);
                a[i + 8] = (int)((r[i] >> (2 * p + 1)) & 0x01010101u);
            }

#define HNET_DO_NT(NT, ACCV)                                                   \
            {                                                                  \
                const unsigned char* f =                                       \
                    lbase + (p * 64 + (NT) * 16 + l15) * LDS_ROW + khalf * 16; \
                v4i b0 = *(const v4i*)(f +  0);                                \
                v4i b1 = *(const v4i*)(f + 32);                                \
                v4i b2 = *(const v4i*)(f + 64);                                \
                v4i b3 = *(const v4i*)(f + 96);                                \
                v16i bf;                                                       \
                _Pragma("unroll")                                              \
                for (int i = 0; i < 4; ++i) {                                  \
                    bf[i] = b0[i]; bf[i + 4] = b1[i];                          \
                    bf[i + 8] = b2[i]; bf[i + 12] = b3[i];                     \
                }                                                              \
                ACCV = __builtin_amdgcn_wmma_f32_16x16x128_fp8_fp8(            \
                    a, bf, (short)0, ACCV, false, false);                      \
            }
            HNET_DO_NT(0, acc0)
            HNET_DO_NT(1, acc1)
            HNET_DO_NT(2, acc2)
            HNET_DO_NT(3, acc3)
#undef HNET_DO_NT
        }
        arow += KTILE;
        __syncthreads();
    }

    const int rbase = mblk * MTILE + wave * 16 + khalf * 8;
    v8f accs[4] = {acc0, acc1, acc2, acc3};
#pragma unroll
    for (int nt = 0; nt < 4; ++nt) {
        const int col = nt * 16 + l15;
#pragma unroll
        for (int v = 0; v < 8; ++v) {
            atomicAdd(&acc[(rbase + v) * N_CMP + col], accs[nt][v] * A_SCALE);
        }
    }
}

__global__ __launch_bounds__(256)
void hnet_min_kernel(const float* __restrict__ acc, int* __restrict__ minslot) {
    int lmin = 0x7F7FFFFF;
    for (int i = blockIdx.x * blockDim.x + threadIdx.x; i < OUT_ELTS;
         i += gridDim.x * blockDim.x)
        lmin = min(lmin, __float_as_int(acc[i]));
#pragma unroll
    for (int off = 16; off > 0; off >>= 1)
        lmin = min(lmin, __shfl_down(lmin, off, 32));
    if ((threadIdx.x & 31) == 0) atomicMin(minslot, lmin);
}

__global__ __launch_bounds__(256)
void hnet_fin_kernel(const float* __restrict__ acc, const int* __restrict__ minslot,
                     float* __restrict__ out) {
    const float mv = __int_as_float(*minslot);
    for (int i = blockIdx.x * blockDim.x + threadIdx.x; i < OUT_ELTS;
         i += gridDim.x * blockDim.x)
        out[i] = acc[i] - mv;
}

extern "C" void kernel_launch(void* const* d_in, const int* in_sizes, int n_in,
                              void* d_out, int out_size, void* d_ws, size_t ws_size,
                              hipStream_t stream) {
    const int*   act  = (const int*)d_in[0];
    const float* B    = (const float*)d_in[1];
    const int*   eidx = (const int*)d_in[2];
    float*       out  = (float*)d_out;

    char* ws = (char*)d_ws;
    unsigned char* type8 = (unsigned char*)(ws + WS_TYPE8);
    unsigned char* b8    = (unsigned char*)(ws + WS_B8);
    float*         acc   = (float*)(ws + WS_ACC);
    int*           mins  = (int*)(ws + WS_MIN);

    hnet_init_kernel<<<(OUT_ELTS + 255) / 256, 256, 0, stream>>>(acc, mins);

    {
        const int total = N_PTS * (E_PAD / 4);
        hnet_types_kernel<<<(total + 255) / 256, 256, 0, stream>>>(act, eidx, type8);
    }
    {
        const int total = N_KTILES * 4096;
        hnet_prepb_kernel<<<(total + 255) / 256, 256, 0, stream>>>(B, b8);
    }
    {
        const int tiles_per_chunk = (N_KTILES + N_KCHUNK - 1) / N_KCHUNK;
        dim3 grid(4, N_KCHUNK);
        hnet_wmma_kernel<<<grid, 256, 0, stream>>>(type8, b8, acc, tiles_per_chunk);
    }

    hnet_min_kernel<<<128, 256, 0, stream>>>(acc, mins);
    hnet_fin_kernel<<<128, 256, 0, stream>>>(acc, mins, out);
}
